// ScaledDotProductAttentionModularized_1202590843689
// MI455X (gfx1250) — matrix operand order repaired and checked by reading, not compiled, not run on hardware
//
#include <hip/hip_runtime.h>

typedef _Float16 v16h __attribute__((ext_vector_type(16)));
typedef _Float16 v8h  __attribute__((ext_vector_type(8)));
typedef _Float16 v4h  __attribute__((ext_vector_type(4)));
typedef _Float16 v2h  __attribute__((ext_vector_type(2)));
typedef __fp16   v2hp __attribute__((ext_vector_type(2)));
typedef float    v8f  __attribute__((ext_vector_type(8)));
typedef float    v4f  __attribute__((ext_vector_type(4)));

#define EMBED   1024
#define NHEADS  16
#define HDIM    64
#define LQ      2048
#define SK      2048
#define BATCH   2
#define ROWSTRIDE (BATCH * EMBED)   // floats between consecutive l (or s) indices

#define BK      64                  // keys processed per iteration
#define WAVES   8
#define QROWS   (WAVES * 16)        // 128 query rows per block

#define KSTR    72                  // halves per row of sK  [BK][HDIM]   (pad 64->72, 16B-aligned rows)
#define VSTR    72                  // halves per row of sVt [HDIM][BK]   (pad 64->72)
#define PSTR    72                  // halves per row of sP  [16][BK] per wave

static __device__ inline v8f wmma_f16(v16h a, v16h b, v8f c) {
    // D = A(16x32 f16) * B(32x16 f16) + C(16x16 f32)
    return __builtin_amdgcn_wmma_f32_16x16x32_f16(
        /*neg_a=*/false, a, /*neg_b=*/false, b,
        /*c_mod=*/(short)0, c, /*reuse_a=*/false, /*reuse_b=*/false);
}

static __device__ inline v16h lds_load16(const _Float16* p) {
    v8h lo = *(const v8h*)p;
    v8h hi = *(const v8h*)(p + 8);
    return __builtin_shufflevector(lo, hi, 0,1,2,3,4,5,6,7,8,9,10,11,12,13,14,15);
}

// packed fp32->f16 conversion (v_cvt_pk_rtz_f16_f32): 2 values per VALU op
static __device__ inline v2h cvt_pk(float a, float b) {
    v2hp t = __builtin_amdgcn_cvt_pkrtz(a, b);
    return __builtin_bit_cast(v2h, t);
}
static __device__ inline v4h pk4(v4f x) {
    v2h lo = cvt_pk(x[0], x[1]);
    v2h hi = cvt_pk(x[2], x[3]);
    return __builtin_shufflevector(lo, hi, 0, 1, 2, 3);
}

// DPP16 row rotate (within each 16-lane row of the wave32): VALU-only cross-lane.
template <int CTRL>
static __device__ inline float dpp_rotf(float x) {
    int r = __builtin_amdgcn_update_dpp(0, __builtin_bit_cast(int, x),
                                        CTRL, 0xF, 0xF, true);
    return __builtin_bit_cast(float, r);
}
// ROW_ROR:n = 0x120 + n. Rotate-reduce over 16 lanes == full row reduction.
static __device__ inline float rowmax16(float x) {
    x = fmaxf(x, dpp_rotf<0x121>(x));
    x = fmaxf(x, dpp_rotf<0x122>(x));
    x = fmaxf(x, dpp_rotf<0x124>(x));
    x = fmaxf(x, dpp_rotf<0x128>(x));
    return x;
}

static __device__ inline float readlane_f(float x, int l) {
    return __builtin_bit_cast(float,
        __builtin_amdgcn_readlane(__builtin_bit_cast(int, x), l));
}

__global__ __launch_bounds__(WAVES * 32)
void fa_fwd_kernel(const float* __restrict__ q,
                   const float* __restrict__ k,
                   const float* __restrict__ v,
                   float* __restrict__ out)
{
    __shared__ __attribute__((aligned(16))) _Float16 sK [BK   * KSTR];
    __shared__ __attribute__((aligned(16))) _Float16 sVt[HDIM * VSTR];
    __shared__ __attribute__((aligned(16))) _Float16 sP [WAVES * 16 * PSTR];

    const int tid    = threadIdx.x;
    const int lane   = tid & 31;
    const int wid    = tid >> 5;
    const int lane15 = lane & 15;
    const int hi     = lane >> 4;        // lane half: 0 or 1
    const int kb     = hi * 8;           // A-layout K base offset for upper lane half

    const int l0      = blockIdx.x * QROWS;
    const int head    = blockIdx.y;      // i = b*H + h
    const int b       = head >> 4;
    const int h       = head & (NHEADS - 1);
    const int baseCol = b * EMBED + h * HDIM;

    // ---------------- Q -> A-matrix fragments (fp32 -> f16) ---------------------
    // Pre-scaled by HEAD_DIM**-0.5 * log2(e): softmax runs in exp2 domain so each
    // exponential is a single v_exp_f32.
    const float qscale = 0.125f * 1.44269504088896340736f;
    const int   qrow   = l0 + wid * 16 + lane15;
    const float* qrp   = q + (size_t)qrow * ROWSTRIDE + baseCol;

    v16h aQ[2];
#pragma unroll
    for (int c = 0; c < 2; ++c) {
        v16h a;
#pragma unroll
        for (int seg = 0; seg < 2; ++seg) {
            const int d0 = c * 32 + hi * 16 + seg * 8;
            v4f f0 = *(const v4f*)(qrp + d0);
            v4f f1 = *(const v4f*)(qrp + d0 + 4);
#pragma unroll
            for (int j = 0; j < 4; ++j) {
                a[seg * 8 + j]     = (_Float16)(f0[j] * qscale);
                a[seg * 8 + 4 + j] = (_Float16)(f1[j] * qscale);
            }
        }
        aQ[c] = a;
    }

    // all-ones B fragment: P @ ones accumulates the softmax denominator into
    // acc[4] on the WMMA pipe (every lane gets the row sum) -- no DPP row-sum.
    v16h onesB;
#pragma unroll
    for (int i = 0; i < 16; ++i) onesB[i] = (_Float16)1.0f;

    // ---------------- running softmax state + output accumulators --------------
    // C/D 16x16 f32 layout: element r of lane holds row M = r + 8*hi, col N = lane&15.
    // acc[0..3] = output d-tiles, acc[4] = softmax denominator l.
    //
    // The running max m is carried IMPLICITLY as a bias fragment mbias = splat(-m)
    // fed to the QK^T WMMA C-operand, so scores arrive pre-subtracted: cS = S - m.
    // Common path (tile max <= m): exp2(cS) directly, no per-element subtract and
    // no overflow possible (cS <= 0). Rare path (tb > 0, ~ln(#tiles) times):
    // subtract tb, rescale acc, shift mbias. m starts at 0 (safe: first tile with
    // any positive score takes the subtract path).
    v8f mbias;
    v8f acc[5];
    const v8f zeroC = {0.f,0.f,0.f,0.f,0.f,0.f,0.f,0.f};
    mbias = zeroC;
#pragma unroll
    for (int t = 0; t < 5; ++t) acc[t] = zeroC;

    _Float16* pw        = &sP[wid * 16 * PSTR];                  // this wave's P scratch
    const _Float16* prd = &sP[(wid * 16 + lane15) * PSTR];       // A-layout readback base

    // ---------------- software-pipelined K/V tile staging -----------------------
    // K: thread owns 4 row-strips (row-major stores).
    const int trow = tid >> 4;            // 0..15
    const int tcol = (tid & 15) * 4;      // 0..60
    // V: thread owns one 4(key)x4(d) block -> register transpose -> b64 stores.
    const int vkey = (tid & 15) * 4;      // 0..60 : key base of block
    const int vcol = (tid >> 4) * 4;      // 0..60 : d   base of block
    v4f kreg[4], vreg[4];

    auto issue_tile_loads = [&](int s0) {
#pragma unroll
        for (int ro = 0; ro < 4; ++ro) {
            kreg[ro] = *(const v4f*)(k + (size_t)(s0 + trow + 16 * ro) * ROWSTRIDE
                                       + baseCol + tcol);
            vreg[ro] = *(const v4f*)(v + (size_t)(s0 + vkey + ro) * ROWSTRIDE
                                       + baseCol + vcol);
        }
    };

    issue_tile_loads(0);

    // ========================== main loop over keys ============================
    for (int s0 = 0; s0 < SK; s0 += BK) {
        __syncthreads();   // previous tile fully consumed by all waves

        // ---- drain staged registers into LDS ----------------------------------
        // K: row-major f16 rows (packed converts).
#pragma unroll
        for (int ro = 0; ro < 4; ++ro)
            *(v4h*)(&sK[(trow + 16 * ro) * KSTR + tcol]) = pk4(kreg[ro]);
        // V: 4x4 register transpose + packed converts, contiguous 8B stores.
#pragma unroll
        for (int j = 0; j < 4; ++j) {
            v2h lo = cvt_pk(vreg[0][j], vreg[1][j]);
            v2h hp = cvt_pk(vreg[2][j], vreg[3][j]);
            *(v4h*)(&sVt[(vcol + j) * VSTR + vkey]) =
                __builtin_shufflevector(lo, hp, 0, 1, 2, 3);
        }
        // issue next tile's global loads now: a full compute phase hides latency
        if (s0 + BK < SK) issue_tile_loads(s0 + BK);
        if (s0 + 2 * BK < SK) {   // and nudge the tile after next toward L2
            __builtin_prefetch(k + (size_t)(s0 + 2 * BK + trow) * ROWSTRIDE + baseCol + tcol, 0, 0);
            __builtin_prefetch(v + (size_t)(s0 + 2 * BK + vkey) * ROWSTRIDE + baseCol + vcol, 0, 0);
        }
        __syncthreads();   // tiles visible to all waves

        // ---- cS = Q @ K^T - m : bias injected through the WMMA C operand --------
        // B fragments hoisted into independent registers so the scheduler can
        // batch all ds_load_b128s and stagger the dscnt waits across WMMAs.
        v8f  cS[4];
        v16h bq[4];
#pragma unroll
        for (int t = 0; t < 4; ++t)
            bq[t] = lds_load16(&sK[(t * 16 + lane15) * KSTR + hi * 16]);
#pragma unroll
        for (int t = 0; t < 4; ++t)
            cS[t] = wmma_f16(aQ[0], bq[t], mbias);                   // d chunk 0
#pragma unroll
        for (int t = 0; t < 4; ++t)
            bq[t] = lds_load16(&sK[(t * 16 + lane15) * KSTR + 32 + hi * 16]);
#pragma unroll
        for (int t = 0; t < 4; ++t)
            cS[t] = wmma_f16(aQ[1], bq[t], cS[t]);                   // d chunk 1

        // ---- wave-shared max test ---------------------------------------------
        float tmax = fmaxf(fmaxf(cS[0][0], cS[1][0]), fmaxf(cS[2][0], cS[3][0]));
#pragma unroll
        for (int r = 1; r < 8; ++r) {
            float a01 = fmaxf(cS[0][r], cS[1][r]);
            float a23 = fmaxf(cS[2][r], cS[3][r]);
            tmax = fmaxf(tmax, fmaxf(a01, a23));
        }
        tmax = rowmax16(tmax);                        // reduce each 16-lane half
        const float tb = fmaxf(readlane_f(tmax, 0),   // combine halves -> uniform
                               readlane_f(tmax, 16));
        if (tb > 0.0f) {                              // scalar-uniform rare path
            const float sc = __builtin_amdgcn_exp2f(-tb);
#pragma unroll
            for (int dt = 0; dt < 5; ++dt) acc[dt] = acc[dt] * sc;
            mbias = mbias - tb;                       // -m' = -m - tb
#pragma unroll
            for (int t = 0; t < 4; ++t)
#pragma unroll
                for (int r = 0; r < 8; ++r) cS[t][r] -= tb;
        }
        // exponentiate (common path: no subtract; cS <= 0 so no overflow)
#pragma unroll
        for (int t = 0; t < 4; ++t)
#pragma unroll
            for (int r = 0; r < 8; ++r)
                cS[t][r] = __builtin_amdgcn_exp2f(cS[t][r]);

        // ---- relayout P: C-frag (cols across lanes) -> A-frag (rows across lanes)
#pragma unroll
        for (int r = 0; r < 8; ++r) {
            const int row = r + 8 * hi;
#pragma unroll
            for (int t = 0; t < 4; ++t)
                pw[row * PSTR + t * 16 + lane15] = (_Float16)cS[t][r];
        }
        // LDS is in-order within a wave; only compiler reordering must be fenced.
        asm volatile("" ::: "memory");

        // ---- O += P @ V (4 d-tiles) and l += P @ 1, per 32-key chunk -----------
        v16h aP[2];
#pragma unroll
        for (int kc = 0; kc < 2; ++kc) {
            v8h lo = *(const v8h*)(prd + kc * 32 + hi * 16);        // K = 16*hi+0..7
            v8h hp = *(const v8h*)(prd + kc * 32 + hi * 16 + 8);    // K = 16*hi+8..15
            aP[kc] = __builtin_shufflevector(lo, hp,
                        0,1,2,3,4,5,6,7,8,9,10,11,12,13,14,15);
        }
#pragma unroll
        for (int kc = 0; kc < 2; ++kc) {
            v16h bv[4];
#pragma unroll
            for (int dt = 0; dt < 4; ++dt)
                bv[dt] = lds_load16(&sVt[(dt * 16 + lane15) * VSTR + kc * 32 + hi * 16]);
#pragma unroll
            for (int dt = 0; dt < 4; ++dt)
                acc[dt] = wmma_f16(aP[kc], bv[dt], acc[dt]);
            acc[4] = wmma_f16(aP[kc], onesB, acc[4]);   // softmax denominator
        }
    }

    // ---------------- epilogue: O = acc / l, fp32, coalesced ---------------------
#pragma unroll
    for (int r = 0; r < 8; ++r) {
        const int row = l0 + wid * 16 + r + 8 * hi;
        const float inv = __builtin_amdgcn_rcpf(acc[4][r]);
        float* orp = out + (size_t)row * ROWSTRIDE + baseCol + lane15;
#pragma unroll
        for (int dt = 0; dt < 4; ++dt)
            orp[dt * 16] = acc[dt][r] * inv;
    }
}

extern "C" void kernel_launch(void* const* d_in, const int* in_sizes, int n_in,
                              void* d_out, int out_size, void* d_ws, size_t ws_size,
                              hipStream_t stream) {
    (void)in_sizes; (void)n_in; (void)d_ws; (void)ws_size; (void)out_size;
    const float* q = (const float*)d_in[0];
    const float* k = (const float*)d_in[1];
    const float* v = (const float*)d_in[2];
    float* out = (float*)d_out;

    dim3 grid(LQ / QROWS, BATCH * NHEADS);  // (16, 32)
    dim3 block(WAVES * 32);                 // 256 threads = 8 waves
    fa_fwd_kernel<<<grid, block, 0, stream>>>(q, k, v, out);
}
